// TimeGAN_88089779241028
// MI455X (gfx1250) — matrix operand order repaired and checked by reading, not compiled, not run on hardware
//
#include <hip/hip_runtime.h>
#include <hip/hip_bf16.h>
#include <math.h>

typedef __attribute__((ext_vector_type(16))) __bf16 v16bf;
typedef __attribute__((ext_vector_type(8)))  float  v8f;

#define GRU_B 256
#define GRU_T 512
#define GRU_F 32
#define GRU_H 128
#define GRU_G 384            // 3*H
#define LD128 136            // padded bf16 row stride for K=128 A tiles
#define LD32  40             // padded bf16 row stride for K=32 A tiles
#define GH_LD 388            // padded f32 row stride for gh staging

__device__ __forceinline__ float sigmoidf_(float x) { return 1.0f / (1.0f + __expf(-x)); }

// A-fragment (16x32 bf16, MxK), contiguous K order matching B:
// lanes 0-15: m=lane, K=kb+0..15
// lanes 16-31: m=lane-16, K=kb+16..31
template <int LD>
__device__ __forceinline__ v16bf load_a_frag(const __bf16* a_lds, int m, int kb, int kh) {
  union { v16bf v; uint4 u[2]; } a;
  const __bf16* base = a_lds + m * LD + kb + 16 * kh;
  a.u[0] = *reinterpret_cast<const uint4*>(base);
  a.u[1] = *reinterpret_cast<const uint4*>(base + 8);
  return a.v;
}

// B-fragment (32x16 bf16, KxN) from pre-converted bf16 weights, B[k][n] = W[n][k]:
// lanes 0-15: n=lane, K=kb+0..15 contiguous; lanes 16-31: n=lane-16, K=kb+16..31.
__device__ __forceinline__ v16bf load_b_frag(const __bf16* __restrict__ W, int ldw,
                                             int n, int kstart) {
  union { v16bf v; uint4 u[2]; } b;
  const __bf16* p = W + (size_t)n * ldw + kstart;
  b.u[0] = *reinterpret_cast<const uint4*>(p);
  b.u[1] = *reinterpret_cast<const uint4*>(p + 8);
  return b.v;
}

// one-shot f32 -> bf16 weight conversion
__global__ __launch_bounds__(256) void cvt_w_bf16(const float* __restrict__ src,
                                                  __bf16* __restrict__ dst, int n) {
  int i = blockIdx.x * 256 + threadIdx.x;
  if (i < n) dst[i] = (__bf16)src[i];
}

// ---------------------------------------------------------------------------
// gx = X @ W_ih^T + b_ih for rows r = t*B + b. Block = 64 rows, 8 waves.
// Wave w owns N-tiles {3w..3w+2} of 24 across 4 M-tiles -> 12 C tiles/wave.
// K=32: X = x [B,T,F] f32 (layer 0).  K=128: X = h_seq bf16 [T*B, H].
// ---------------------------------------------------------------------------
template <int K, int LD>
__global__ __launch_bounds__(256) void gru_inproj_wmma(
    const float* __restrict__ Xf, const __bf16* __restrict__ Xb,
    const __bf16* __restrict__ Wb, const float* __restrict__ bias,
    float* __restrict__ gx)
{
  __shared__ __attribute__((aligned(16))) __bf16 a_lds[64 * LD];
  const int tid = threadIdx.x;
  const int r0  = blockIdx.x * 64;

  if (K == 32) {                         // layer 0: f32 x, strided [B,T,F]
    for (int idx = tid; idx < 64 * 32; idx += 256) {
      int m = idx >> 5, k = idx & 31;
      int r = r0 + m;
      int b = r & (GRU_B - 1), t = r >> 8;
      a_lds[m * LD + k] = (__bf16)Xf[((size_t)b * GRU_T + t) * GRU_F + k];
    }
  } else {                               // bf16 h_seq, contiguous rows
    for (int idx = tid; idx < 64 * 16; idx += 256) {
      int m = idx >> 4, c = idx & 15;
      *reinterpret_cast<uint4*>(&a_lds[m * LD + c * 8]) =
          *reinterpret_cast<const uint4*>(&Xb[(size_t)(r0 + m) * GRU_H + c * 8]);
    }
  }
  __syncthreads();

  const int w = tid >> 5, lane = tid & 31, kh = lane >> 4, nl = lane & 15;
  const v8f zero = {0.f, 0.f, 0.f, 0.f, 0.f, 0.f, 0.f, 0.f};

  v8f acc[4][3];
#pragma unroll
  for (int mt = 0; mt < 4; ++mt)
#pragma unroll
    for (int i = 0; i < 3; ++i) acc[mt][i] = zero;

#pragma unroll
  for (int kt = 0; kt < K / 32; ++kt) {
    const int kb = kt * 32;
    v16bf afr[4];
#pragma unroll
    for (int mt = 0; mt < 4; ++mt)
      afr[mt] = load_a_frag<LD>(a_lds, mt * 16 + nl, kb, kh);
    v16bf bfr[3];
#pragma unroll
    for (int i = 0; i < 3; ++i)
      bfr[i] = load_b_frag(Wb, K, (w * 3 + i) * 16 + nl, kb + kh * 16);
#pragma unroll
    for (int mt = 0; mt < 4; ++mt)
#pragma unroll
      for (int i = 0; i < 3; ++i)
        acc[mt][i] = __builtin_amdgcn_wmma_f32_16x16x32_bf16(
            false, afr[mt], false, bfr[i], (short)0, acc[mt][i], false, false);
  }

  const int mb = 8 * kh;
#pragma unroll
  for (int i = 0; i < 3; ++i) {
    const int ncol = (w * 3 + i) * 16 + nl;
    const float bv = bias[ncol];
#pragma unroll
    for (int mt = 0; mt < 4; ++mt)
#pragma unroll
      for (int v = 0; v < 8; ++v)
        gx[(size_t)(r0 + mt * 16 + mb + v) * GRU_G + ncol] = acc[mt][i][v] + bv;
  }
}

// ---------------------------------------------------------------------------
// Sequential GRU scan. Block = 16 batch rows, 384 threads = 12 waves.
// Wave w owns N-tiles {2w, 2w+1} of 24; W_hh bf16 fragments persist in
// registers across all T steps. h kept in LDS (f32 master + bf16 A-layout).
// ---------------------------------------------------------------------------
__global__ __launch_bounds__(384) void gru_scan_wmma(
    const float* __restrict__ gx, const __bf16* __restrict__ Whh,
    const float* __restrict__ bhh, __bf16* __restrict__ hseq)
{
  __shared__ __attribute__((aligned(16))) __bf16 h_bf[16 * LD128];
  __shared__ float h_f[16 * GRU_H];
  __shared__ float gh[16 * GH_LD];

  const int tid  = threadIdx.x;
  const int b0   = blockIdx.x * 16;
  const int w    = tid >> 5;
  const int lane = tid & 31;
  const int kh   = lane >> 4;
  const int nl   = lane & 15;

  // Persistent B fragments (W_hh^T), 64 VGPRs per lane, loaded once.
  v16bf bfr[2][4];
#pragma unroll
  for (int i = 0; i < 2; ++i) {
    const int n0 = (w * 2 + i) * 16;
#pragma unroll
    for (int kt = 0; kt < 4; ++kt)
      bfr[i][kt] = load_b_frag(Whh, GRU_H, n0 + nl, kt * 32 + kh * 16);
  }

  for (int idx = tid; idx < 16 * GRU_H; idx += 384) {
    h_f[idx] = 0.0f;
    int m = idx >> 7, j = idx & 127;
    h_bf[m * LD128 + j] = (__bf16)0.0f;
  }
  __syncthreads();

  for (int t = 0; t < GRU_T; ++t) {
    // gh = h @ W_hh^T (bf16 WMMA, f32 accumulate), staged to LDS
#pragma unroll
    for (int i = 0; i < 2; ++i) {
      v8f c = {0.f, 0.f, 0.f, 0.f, 0.f, 0.f, 0.f, 0.f};
#pragma unroll
      for (int kt = 0; kt < 4; ++kt) {
        v16bf a = load_a_frag<LD128>(h_bf, nl, kt * 32, kh);
        c = __builtin_amdgcn_wmma_f32_16x16x32_bf16(false, a, false, bfr[i][kt],
                                                    (short)0, c, false, false);
      }
      const int ncol = (w * 2 + i) * 16 + nl;
      const int mb   = 8 * kh;
#pragma unroll
      for (int v = 0; v < 8; ++v) gh[(mb + v) * GH_LD + ncol] = c[v];
    }
    __syncthreads();

    // fused gate update
    const float* gxt = gx + ((size_t)t * GRU_B + b0) * GRU_G;
    for (int idx = tid; idx < 16 * GRU_H; idx += 384) {
      const int bl = idx >> 7, j = idx & 127;
      const float* gxr = gxt + (size_t)bl * GRU_G;
      float hr = gh[bl * GH_LD + j]       + bhh[j];
      float hz = gh[bl * GH_LD + j + 128] + bhh[j + 128];
      float hn = gh[bl * GH_LD + j + 256] + bhh[j + 256];
      float r  = sigmoidf_(gxr[j] + hr);
      float z  = sigmoidf_(gxr[j + 128] + hz);
      float n  = tanhf(gxr[j + 256] + r * hn);
      float hp = h_f[idx];
      float hnew = (1.0f - z) * n + z * hp;
      h_f[idx] = hnew;
      h_bf[bl * LD128 + j] = (__bf16)hnew;
      hseq[((size_t)t * GRU_B + b0 + bl) * GRU_H + j] = (__bf16)hnew;
      if (t + 1 < GRU_T)
        __builtin_prefetch(gxr + (size_t)GRU_B * GRU_G, 0, 0);  // next step's gx
    }
    __syncthreads();
  }
}

// ---------------------------------------------------------------------------
// out[b][t][:] = sigmoid(h @ W_out^T + b_out). Block = 64 rows, 8 waves.
// Wave w -> N-tile w (N=128) across 4 M-tiles.
// ---------------------------------------------------------------------------
__global__ __launch_bounds__(256) void gru_out_wmma(
    const __bf16* __restrict__ hs, const __bf16* __restrict__ Wo,
    const float* __restrict__ bo, float* __restrict__ out)
{
  __shared__ __attribute__((aligned(16))) __bf16 a_lds[64 * LD128];
  const int tid = threadIdx.x;
  const int r0  = blockIdx.x * 64;
  for (int idx = tid; idx < 64 * 16; idx += 256) {
    int m = idx >> 4, c = idx & 15;
    *reinterpret_cast<uint4*>(&a_lds[m * LD128 + c * 8]) =
        *reinterpret_cast<const uint4*>(&hs[(size_t)(r0 + m) * GRU_H + c * 8]);
  }
  __syncthreads();

  const int w = tid >> 5, lane = tid & 31, kh = lane >> 4, nl = lane & 15;
  const v8f zero = {0.f, 0.f, 0.f, 0.f, 0.f, 0.f, 0.f, 0.f};
  const int n0 = w * 16;

  v8f acc[4];
#pragma unroll
  for (int mt = 0; mt < 4; ++mt) acc[mt] = zero;

#pragma unroll
  for (int kt = 0; kt < 4; ++kt) {
    const int kb = kt * 32;
    v16bf b = load_b_frag(Wo, GRU_H, n0 + nl, kb + kh * 16);
#pragma unroll
    for (int mt = 0; mt < 4; ++mt) {
      v16bf a = load_a_frag<LD128>(a_lds, mt * 16 + nl, kb, kh);
      acc[mt] = __builtin_amdgcn_wmma_f32_16x16x32_bf16(false, a, false, b,
                                                        (short)0, acc[mt], false, false);
    }
  }

  const int ncol = n0 + nl;
  const int mb   = 8 * kh;
  const float bv = bo[ncol];
#pragma unroll
  for (int mt = 0; mt < 4; ++mt)
#pragma unroll
    for (int v = 0; v < 8; ++v) {
      const int r = r0 + mt * 16 + mb + v;
      const int b = r & (GRU_B - 1);
      const int t = r >> 8;
      out[((size_t)b * GRU_T + t) * GRU_H + ncol] = sigmoidf_(acc[mt][v] + bv);
    }
}

extern "C" void kernel_launch(void* const* d_in, const int* in_sizes, int n_in,
                              void* d_out, int out_size, void* d_ws, size_t ws_size,
                              hipStream_t stream) {
  (void)in_sizes; (void)n_in; (void)out_size; (void)ws_size;
  const float* x     = (const float*)d_in[0];  // [B,T,F]
  const float* W_ih0 = (const float*)d_in[1];  // [384,32]
  const float* W_ihr = (const float*)d_in[2];  // [2,384,128]
  const float* W_hh  = (const float*)d_in[3];  // [3,384,128]
  const float* b_ih  = (const float*)d_in[4];  // [3,384]
  const float* b_hh  = (const float*)d_in[5];  // [3,384]
  const float* W_out = (const float*)d_in[6];  // [128,128]
  const float* b_out = (const float*)d_in[7];  // [128]
  float* out = (float*)d_out;                  // [B,T,H]

  char* ws = (char*)d_ws;
  size_t off = 0;
  float*  gx = (float*)(ws + off);  off += (size_t)GRU_T * GRU_B * GRU_G * 4;   // [T*B,384] f32
  __bf16* hA = (__bf16*)(ws + off); off += (size_t)GRU_T * GRU_B * GRU_H * 2;   // [T*B,128] bf16
  __bf16* hB = (__bf16*)(ws + off); off += (size_t)GRU_T * GRU_B * GRU_H * 2;
  __bf16* Wb_ih0 = (__bf16*)(ws + off); off += (size_t)GRU_G * GRU_F * 2;
  __bf16* Wb_ihr = (__bf16*)(ws + off); off += (size_t)2 * GRU_G * GRU_H * 2;
  __bf16* Wb_hh  = (__bf16*)(ws + off); off += (size_t)3 * GRU_G * GRU_H * 2;
  __bf16* Wb_out = (__bf16*)(ws + off); off += (size_t)GRU_H * GRU_H * 2;

  // one-shot weight conversions (tiny)
  {
    int n0 = GRU_G * GRU_F, n1 = 2 * GRU_G * GRU_H, n2 = 3 * GRU_G * GRU_H, n3 = GRU_H * GRU_H;
    cvt_w_bf16<<<(n0 + 255) / 256, 256, 0, stream>>>(W_ih0, Wb_ih0, n0);
    cvt_w_bf16<<<(n1 + 255) / 256, 256, 0, stream>>>(W_ihr, Wb_ihr, n1);
    cvt_w_bf16<<<(n2 + 255) / 256, 256, 0, stream>>>(W_hh, Wb_hh, n2);
    cvt_w_bf16<<<(n3 + 255) / 256, 256, 0, stream>>>(W_out, Wb_out, n3);
  }

  const int R = GRU_T * GRU_B;
  dim3 gIn(R / 64), bIn(256);
  dim3 gScan(GRU_B / 16), bScan(384);

  // layer 0
  gru_inproj_wmma<32, LD32><<<gIn, bIn, 0, stream>>>(x, (const __bf16*)nullptr, Wb_ih0, b_ih, gx);
  gru_scan_wmma<<<gScan, bScan, 0, stream>>>(gx, Wb_hh, b_hh, hA);
  // layer 1
  gru_inproj_wmma<128, LD128><<<gIn, bIn, 0, stream>>>((const float*)nullptr, hA, Wb_ihr, b_ih + GRU_G, gx);
  gru_scan_wmma<<<gScan, bScan, 0, stream>>>(gx, Wb_hh + GRU_G * GRU_H, b_hh + GRU_G, hB);
  // layer 2
  gru_inproj_wmma<128, LD128><<<gIn, bIn, 0, stream>>>((const float*)nullptr, hB, Wb_ihr + GRU_G * GRU_H, b_ih + 2 * GRU_G, gx);
  gru_scan_wmma<<<gScan, bScan, 0, stream>>>(gx, Wb_hh + 2 * GRU_G * GRU_H, b_hh + 2 * GRU_G, hA);
  // output head
  gru_out_wmma<<<gIn, bIn, 0, stream>>>(hA, Wb_out, b_out, out);
}
